// LSTM_45947560133377
// MI455X (gfx1250) — compile-verified, matrix operand order checked by reading, not run on hardware
//
#include <hip/hip_runtime.h>
#include <hip/hip_bf16.h>
#include <stdint.h>

// ---------------------------------------------------------------------------
// MI455X (gfx1250) persistent-LSTM implementation.
//  - bf16 WMMA (v_wmma_f32_16x16x32_bf16) for every matmul
//  - fused [W_ih1 | W_hh1] (512x256 bf16 = 256KB) resident in LDS all T steps
//  - hidden state ping-pong in LDS, cell state in registers
//  - 32 workgroups x 32 batch rows; 8 waves each own 16 gate columns across
//    all four gate sections -> LSTM pointwise update is register-only.
// ---------------------------------------------------------------------------

typedef __bf16 bf16;
typedef __attribute__((ext_vector_type(16))) __bf16   bf16x16;
typedef __attribute__((ext_vector_type(8)))  float    floatx8;
typedef __attribute__((ext_vector_type(4)))  uint32_t uint32x4;

#define BATCH   1024
#define TLEN    512
#define HID     128
#define GATES   512     // 4*HID
#define BM      32      // batch rows per workgroup
#define D0PAD   64      // 36-dim input padded to 64 (two K=32 WMMA steps)
#define ABSIN   10000
#define MAXD    2048    // (4097-1)/2
#define NPD     4097
#define NBLK    32

// ---- workspace layout (bytes, all 256B aligned) ---------------------------
#define OFF_WC0X   0u                          // [512][64]  bf16  ( 64KB)
#define OFF_WC0H   65536u                      // [512][128] bf16  (128KB)
#define OFF_WC1    196608u                     // [512][256] bf16  (256KB)
#define OFF_WPD    458752u                     // [4097][128] bf16 (~1MB)
#define OFF_WBLK   1507584u                    // [32][128]  bf16
#define OFF_B0     1515776u                    // [512] f32 (b_ih0+b_hh0)
#define OFF_B1     1517824u                    // [512] f32
#define OFF_HFIN   1519872u                    // [1024][128] bf16 final h

union FragU {
    bf16x16  v;
    uint32x4 q[2];
};

// 16-bit WMMA operand fragment from a row-major matrix (stride in elements).
// lane&15 selects the striped row (A: M-row, B: N-column); half = lane>>4
// selects the K+8 half per ISA 7.12.2 (VGPR0..3 = K 0..7 / 8..15, VGPR4..7 =
// K 16..23 / 24..31).  Two 16-byte loads per fragment.
__device__ inline bf16x16 load_frag16(const bf16* base, int row, int stride,
                                      int k0, int half) {
    FragU f;
    const bf16* p = base + row * stride + k0 + (half << 3);
    f.q[0] = *(const uint32x4*)(p);
    f.q[1] = *(const uint32x4*)(p + 16);
    return f.v;
}

__device__ inline floatx8 wmma_bf16(bf16x16 a, bf16x16 b, floatx8 c) {
    return __builtin_amdgcn_wmma_f32_16x16x32_bf16(
        /*neg_a=*/false, a, /*neg_b=*/false, b,
        /*c_mod=*/(short)0, c, /*reuse_a=*/false, /*reuse_b=*/false);
}

__device__ inline float sigf(float x) { return 1.0f / (1.0f + __expf(-x)); }

// ---------------------------------------------------------------------------
// Kernel 1: convert / fuse weights to bf16 in workspace
// ---------------------------------------------------------------------------
__global__ void convert_weights(const float* __restrict__ Wih0,  // [512][36]
                                const float* __restrict__ Whh0,  // [512][128]
                                const float* __restrict__ Wih1,  // [512][128]
                                const float* __restrict__ Whh1,  // [512][128]
                                const float* __restrict__ bih0,
                                const float* __restrict__ bhh0,
                                const float* __restrict__ bih1,
                                const float* __restrict__ bhh1,
                                const float* __restrict__ Wpdf,  // [4097][128]
                                const float* __restrict__ Wblkf, // [32][128]
                                char* __restrict__ ws) {
    bf16*  Wc0x = (bf16*)(ws + OFF_WC0X);
    bf16*  Wc0h = (bf16*)(ws + OFF_WC0H);
    bf16*  Wc1  = (bf16*)(ws + OFF_WC1);
    bf16*  Wpd  = (bf16*)(ws + OFF_WPD);
    bf16*  Wblk = (bf16*)(ws + OFF_WBLK);
    float* b0   = (float*)(ws + OFF_B0);
    float* b1   = (float*)(ws + OFF_B1);

    int i = blockIdx.x * 256 + threadIdx.x;
    if (i < 512 * 64) {                       // W_ih0 padded 36 -> 64
        int n = i >> 6, k = i & 63;
        Wc0x[i] = (bf16)(k < 36 ? Wih0[n * 36 + k] : 0.0f);
        return;
    }
    i -= 512 * 64;
    if (i < 512 * 128) { Wc0h[i] = (bf16)Whh0[i]; return; }
    i -= 512 * 128;
    if (i < 512 * 256) {                      // fused [W_ih1 | W_hh1]
        int n = i >> 8, k = i & 255;
        Wc1[i] = (bf16)(k < 128 ? Wih1[n * 128 + k] : Whh1[n * 128 + k - 128]);
        return;
    }
    i -= 512 * 256;
    if (i < NPD * 128) { Wpd[i] = (bf16)Wpdf[i]; return; }
    i -= NPD * 128;
    if (i < NBLK * 128) { Wblk[i] = (bf16)Wblkf[i]; return; }
    i -= NBLK * 128;
    if (i < 512) { b0[i] = bih0[i] + bhh0[i]; return; }
    i -= 512;
    if (i < 512) { b1[i] = bih1[i] + bhh1[i]; return; }
}
#define CVT_TOTAL (512*64 + 512*128 + 512*256 + NPD*128 + NBLK*128 + 512 + 512)

// ---------------------------------------------------------------------------
// Kernel 2: persistent 2-layer LSTM scan (embedding fused per-step)
// ---------------------------------------------------------------------------
__global__ __launch_bounds__(256, 1)
void lstm_persistent(const int*   __restrict__ seq,        // [B][T][2]
                     const float* __restrict__ abs_emb,    // [10000][24]
                     const float* __restrict__ delta_emb,  // [4097][24]
                     const float* __restrict__ block_emb,  // [32][12]
                     const char*  __restrict__ ws_ro,
                     char*        __restrict__ ws) {
    const bf16*  Wc0x = (const bf16*)(ws_ro + OFF_WC0X);
    const bf16*  Wc0h = (const bf16*)(ws_ro + OFF_WC0H);
    const bf16*  Wc1g = (const bf16*)(ws_ro + OFF_WC1);
    const float* b0g  = (const float*)(ws_ro + OFF_B0);
    const float* b1g  = (const float*)(ws_ro + OFF_B1);
    bf16*        hfin = (bf16*)(ws + OFF_HFIN);

    // 292KB static LDS: OK on gfx1250 (320KB/WGP); forces 1 WG per WGP.
    __shared__ __align__(16) bf16 sWc1[GATES * 256];  // 256KB fused layer-1 W
    __shared__ __align__(16) bf16 sH0[2][BM][HID];    //  16KB ping-pong
    __shared__ __align__(16) bf16 sH1[2][BM][HID];    //  16KB ping-pong
    __shared__ __align__(16) bf16 sX[BM][D0PAD];      //   4KB embedded input

    const int tid    = threadIdx.x;
    const int lane   = tid & 31;
    const int wave   = tid >> 5;      // 0..7 : owns gate columns 16w..16w+15
    const int rowsel = lane & 15;
    const int half   = lane >> 4;
    const int wgBase = blockIdx.x * BM;

    // preload fused layer-1 weights into LDS (L2 -> LDS, once)
    {
        const uint32x4* src = (const uint32x4*)Wc1g;
        uint32x4*       dst = (uint32x4*)sWc1;
        for (int i = tid; i < (GATES * 256 * 2) / 16; i += 256) dst[i] = src[i];
    }
    for (int i = tid; i < 2 * BM * HID; i += 256) {
        ((bf16*)sH0)[i] = (bf16)0.0f;
        ((bf16*)sH1)[i] = (bf16)0.0f;
    }

    // per-lane gate biases (column = q*128 + wave*16 + rowsel)
    float bias0[4], bias1[4];
#pragma unroll
    for (int q = 0; q < 4; ++q) {
        int n = q * 128 + wave * 16 + rowsel;
        bias0[q] = b0g[n];
        bias1[q] = b1g[n];
    }

    // cell state in registers: rows (mt*16 + 8*half + v), col wave*16+rowsel
    float c0[2][8] = {}, c1[2][8] = {};

    __syncthreads();

    const floatx8 z8 = {0.f, 0.f, 0.f, 0.f, 0.f, 0.f, 0.f, 0.f};

    for (int t = 0; t < TLEN; ++t) {
        const int wb = t & 1, rb = wb ^ 1;

        // ---- stage embedded x_t (BM x 64, cols 36..63 zero) ----
        for (int i = tid; i < BM * D0PAD; i += 256) {
            int r = i >> 6, col = i & 63;
            int gb = wgBase + r;
            float val = 0.0f;
            if (col < 24) {
                const float* src;
                if (t == 0) {
                    int p = seq[(gb * TLEN) * 2] % ABSIN;
                    src = abs_emb + p * 24;
                } else {
                    int d = seq[(gb * TLEN + t) * 2] - seq[(gb * TLEN + t - 1) * 2];
                    d = (d < -MAXD ? -MAXD : (d > MAXD ? MAXD : d)) + MAXD;
                    src = delta_emb + d * 24;
                }
                val = src[col];
            } else if (col < 36) {
                int bk = seq[(gb * TLEN + t) * 2 + 1];
                val = block_emb[bk * 12 + (col - 24)];
            }
            sX[r][col] = (bf16)val;
        }
        __syncthreads();

        // ---- layer 0: g = [x_t | h0_prev] * [Wc0x | Wc0h]^T   (K = 64+128)
        floatx8 acc[2][4];
#pragma unroll
        for (int mt = 0; mt < 2; ++mt)
#pragma unroll
            for (int q = 0; q < 4; ++q) acc[mt][q] = z8;

#pragma unroll
        for (int kt = 0; kt < 6; ++kt) {
            bf16x16 a[2];
            if (kt < 2) {
                a[0] = load_frag16(&sX[0][0], rowsel,      D0PAD, kt * 32, half);
                a[1] = load_frag16(&sX[0][0], rowsel + 16, D0PAD, kt * 32, half);
            } else {
                a[0] = load_frag16(&sH0[rb][0][0], rowsel,      HID, (kt - 2) * 32, half);
                a[1] = load_frag16(&sH0[rb][0][0], rowsel + 16, HID, (kt - 2) * 32, half);
            }
#pragma unroll
            for (int q = 0; q < 4; ++q) {
                int n = q * 128 + wave * 16 + rowsel;
                bf16x16 b = (kt < 2)
                    ? load_frag16(Wc0x, n, D0PAD, kt * 32, half)        // L2
                    : load_frag16(Wc0h, n, HID, (kt - 2) * 32, half);   // L2
                acc[0][q] = wmma_bf16(a[0], b, acc[0][q]);
                acc[1][q] = wmma_bf16(a[1], b, acc[1][q]);
            }
        }

        // pointwise LSTM update, layer 0 (register-only gates)
#pragma unroll
        for (int mt = 0; mt < 2; ++mt)
#pragma unroll
            for (int v = 0; v < 8; ++v) {
                float gi = sigf(acc[mt][0][v] + bias0[0]);
                float gf = sigf(acc[mt][1][v] + bias0[1]);
                float gg = tanhf(acc[mt][2][v] + bias0[2]);
                float go = sigf(acc[mt][3][v] + bias0[3]);
                float c  = gf * c0[mt][v] + gi * gg;
                c0[mt][v] = c;
                float h  = go * tanhf(c);
                sH0[wb][mt * 16 + half * 8 + v][wave * 16 + rowsel] = (bf16)h;
            }
        __syncthreads();

        // ---- layer 1: g = [h0_t | h1_prev] * [W_ih1 | W_hh1]^T (K = 256, LDS)
#pragma unroll
        for (int mt = 0; mt < 2; ++mt)
#pragma unroll
            for (int q = 0; q < 4; ++q) acc[mt][q] = z8;

#pragma unroll
        for (int kt = 0; kt < 8; ++kt) {
            const bf16* asrc = (kt < 4) ? &sH0[wb][0][0] : &sH1[rb][0][0];
            int k0 = (kt & 3) * 32;
            bf16x16 a[2];
            a[0] = load_frag16(asrc, rowsel,      HID, k0, half);
            a[1] = load_frag16(asrc, rowsel + 16, HID, k0, half);
#pragma unroll
            for (int q = 0; q < 4; ++q) {
                int n = q * 128 + wave * 16 + rowsel;
                bf16x16 b = load_frag16(sWc1, n, 256, kt * 32, half);   // LDS
                acc[0][q] = wmma_bf16(a[0], b, acc[0][q]);
                acc[1][q] = wmma_bf16(a[1], b, acc[1][q]);
            }
        }

        // pointwise LSTM update, layer 1
#pragma unroll
        for (int mt = 0; mt < 2; ++mt)
#pragma unroll
            for (int v = 0; v < 8; ++v) {
                float gi = sigf(acc[mt][0][v] + bias1[0]);
                float gf = sigf(acc[mt][1][v] + bias1[1]);
                float gg = tanhf(acc[mt][2][v] + bias1[2]);
                float go = sigf(acc[mt][3][v] + bias1[3]);
                float c  = gf * c1[mt][v] + gi * gg;
                c1[mt][v] = c;
                float h  = go * tanhf(c);
                int rrow = mt * 16 + half * 8 + v;
                sH1[wb][rrow][wave * 16 + rowsel] = (bf16)h;
                if (t == TLEN - 1)
                    hfin[(wgBase + rrow) * HID + wave * 16 + rowsel] = (bf16)h;
            }
        __syncthreads();
    }
}

// ---------------------------------------------------------------------------
// Kernel 3: output projections via WMMA; one 16x16 tile per wave.
// Columns 0..4096 -> page_delta_pred, 4097..4128 -> block_pred.
// ---------------------------------------------------------------------------
#define NTOT  (NPD + NBLK)                 // 4129
#define NTILE ((NTOT + 15) / 16)           // 259
#define MTILE (BATCH / 16)                 // 64

__global__ __launch_bounds__(256, 2)
void final_proj(const char*  __restrict__ ws_ro,
                const float* __restrict__ b_pd,
                const float* __restrict__ b_blk,
                float*       __restrict__ out) {
    const bf16* hf   = (const bf16*)(ws_ro + OFF_HFIN);
    const bf16* Wpd  = (const bf16*)(ws_ro + OFF_WPD);
    const bf16* Wblk = (const bf16*)(ws_ro + OFF_WBLK);

    int gwid = blockIdx.x * 8 + (threadIdx.x >> 5);
    int mt = gwid / NTILE, nt = gwid % NTILE;
    if (mt >= MTILE) return;

    int lane   = threadIdx.x & 31;
    int rowsel = lane & 15;
    int half   = lane >> 4;
    int n      = nt * 16 + rowsel;

    const bf16* wsrc;
    float bias = 0.0f;
    if (n < NPD)            { wsrc = Wpd  + n * HID;         bias = b_pd[n]; }
    else if (n < NTOT)      { wsrc = Wblk + (n - NPD) * HID; bias = b_blk[n - NPD]; }
    else                    { wsrc = Wblk; }   // padding lane, store skipped

    floatx8 acc = {0.f, 0.f, 0.f, 0.f, 0.f, 0.f, 0.f, 0.f};
#pragma unroll
    for (int kt = 0; kt < 4; ++kt) {
        bf16x16 a = load_frag16(hf, mt * 16 + rowsel, HID, kt * 32, half);
        bf16x16 b = load_frag16(wsrc, 0, HID, kt * 32, half);
        acc = wmma_bf16(a, b, acc);
    }
#pragma unroll
    for (int v = 0; v < 8; ++v) {
        int m = mt * 16 + half * 8 + v;
        float val = acc[v] + bias;
        if (n < NPD)       out[m * NPD + n] = val;
        else if (n < NTOT) out[BATCH * NPD + m * NBLK + (n - NPD)] = val;
    }
}

// ---------------------------------------------------------------------------
extern "C" void kernel_launch(void* const* d_in, const int* in_sizes, int n_in,
                              void* d_out, int out_size, void* d_ws, size_t ws_size,
                              hipStream_t stream) {
    const int*   seq       = (const int*)  d_in[0];
    const float* abs_emb   = (const float*)d_in[1];
    const float* delta_emb = (const float*)d_in[2];
    const float* block_emb = (const float*)d_in[3];
    const float* W_ih0     = (const float*)d_in[4];
    const float* W_hh0     = (const float*)d_in[5];
    const float* b_ih0     = (const float*)d_in[6];
    const float* b_hh0     = (const float*)d_in[7];
    const float* W_ih1     = (const float*)d_in[8];
    const float* W_hh1     = (const float*)d_in[9];
    const float* b_ih1     = (const float*)d_in[10];
    const float* b_hh1     = (const float*)d_in[11];
    const float* W_pd      = (const float*)d_in[12];
    const float* b_pd      = (const float*)d_in[13];
    const float* W_blk     = (const float*)d_in[14];
    const float* b_blk     = (const float*)d_in[15];
    float* out = (float*)d_out;
    char*  ws  = (char*)d_ws;

    int cvt_blocks = (CVT_TOTAL + 255) / 256;
    convert_weights<<<cvt_blocks, 256, 0, stream>>>(
        W_ih0, W_hh0, W_ih1, W_hh1, b_ih0, b_hh0, b_ih1, b_hh1, W_pd, W_blk, ws);

    lstm_persistent<<<BATCH / BM, 256, 0, stream>>>(
        seq, abs_emb, delta_emb, block_emb, ws, ws);

    int proj_blocks = (MTILE * NTILE + 7) / 8;
    final_proj<<<proj_blocks, 256, 0, stream>>>(ws, b_pd, b_blk, out);
}
